// Pseudo3DConv_15951508537870
// MI455X (gfx1250) — matrix operand order repaired and checked by reading, not compiled, not run on hardware
//
#include <hip/hip_runtime.h>
#include <hip/hip_bf16.h>

// CDNA5 / gfx1250: wave32, WMMA 16x16x32 f16 -> f32, async LDS staging
typedef __attribute__((ext_vector_type(16))) _Float16 v16h;
typedef __attribute__((ext_vector_type(8)))  float    v8f;
typedef __attribute__((ext_vector_type(4)))  int      v4i;

#define NPT 16384
#define NBK 16

struct alignas(16) u4 { unsigned int a, b, c, d; };
union U16 { v16h h; u4 u[2]; };

__device__ __forceinline__ float lrelu(float x) { return x > 0.f ? x : 0.01f * x; }

__device__ __forceinline__ v8f wmma16(v16h a, v16h b, v8f c) {
  return __builtin_amdgcn_wmma_f32_16x16x32_f16(false, a, false, b, (short)0, c, false, false);
}

// ---- CDNA5 async global->LDS copy (16B per lane), tracked by ASYNCcnt ----
typedef __attribute__((address_space(1))) v4i* gv4i_p;
typedef __attribute__((address_space(3))) v4i* lv4i_p;

__device__ __forceinline__ void async_cp16(void* g, void* l) {
#if __has_builtin(__builtin_amdgcn_global_load_async_to_lds_b128)
  __builtin_amdgcn_global_load_async_to_lds_b128((gv4i_p)g, (lv4i_p)l, 0, 0);
#else
  asm volatile("global_load_async_to_lds_b128 %0, %1, off"
               :: "v"((unsigned)(unsigned long long)l),
                  "v"((unsigned long long)g)
               : "memory");
#endif
}
__device__ __forceinline__ void wait_async0() {
#if __has_builtin(__builtin_amdgcn_s_wait_asynccnt)
  __builtin_amdgcn_s_wait_asynccnt(0);
#else
  asm volatile("s_wait_asynccnt 0x0" ::: "memory");
#endif
}
// compiler-only reordering fence (same-wave LDS ops complete in order in HW)
__device__ __forceinline__ void sched_fence() { asm volatile("" ::: "memory"); }

// Load an A fragment: two contiguous 8-half groups (16B each).
__device__ __forceinline__ v16h load_a2(const _Float16* p0, const _Float16* p1) {
  U16 t;
  t.u[0] = *reinterpret_cast<const u4*>(p0);
  t.u[1] = *reinterpret_cast<const u4*>(p1);
  return t.h;
}

// Load a packed B fragment: lane reads its own 32 contiguous bytes.
__device__ __forceinline__ v16h load_bfrag(const _Float16* frag, int lane) {
  const u4* q = reinterpret_cast<const u4*>(frag) + (lane << 1);
  U16 t; t.u[0] = q[0]; t.u[1] = q[1];
  return t.h;
}

// -------- Pack weight matrix W[out_f][in_f] (f32) into WMMA B fragments (f16).
__global__ void pack_w(const float* __restrict__ W, _Float16* __restrict__ dst,
                       int in_f, int NT, int KC) {
  int t = blockIdx.x * blockDim.x + threadIdx.x;
  int total = KC * NT * 512;
  if (t >= total) return;
  int e    = t & 15;
  int lane = (t >> 4) & 31;
  int f    = t >> 9;
  int nt = f % NT, kc = f / NT;
  int n = nt * 16 + (lane & 15);
  int k = kc * 32 + ((e >> 3) << 4) + ((lane >> 4) << 3) + (e & 7);
  dst[t] = (_Float16)W[n * in_f + k];
}

// -------- img_feat [32][N] -> imf16 [N][32] + lrelu into final cols 128..159
__global__ void prep_imf(const float* __restrict__ img, _Float16* __restrict__ imf16,
                         _Float16* __restrict__ fbuf) {
  const int n = blockIdx.x * blockDim.x + threadIdx.x;
  if (n >= NPT) return;
#pragma unroll
  for (int c = 0; c < 32; ++c) {
    float v = img[c * NPT + n];
    imf16[n * 32 + c] = (_Float16)v;
    fbuf[n * 416 + 128 + c] = (_Float16)lrelu(v);
  }
}

// -------- point MLP: cf = lrelu(pts@W1^T + b1)@W2^T + b2   (3->64->128)
__global__ void __launch_bounds__(256)
point_mlp(const float* __restrict__ cloud,
          const float* __restrict__ w1, const float* __restrict__ b1,
          const float* __restrict__ w2, const float* __restrict__ b2,
          _Float16* __restrict__ cf16, _Float16* __restrict__ fbuf) {
  __shared__ float sw2[128 * 64];
  __shared__ float sw1[64 * 3];
  for (int i = threadIdx.x; i < 128 * 64; i += 256) sw2[i] = w2[i];
  for (int i = threadIdx.x; i < 64 * 3; i += 256) sw1[i] = w1[i];
  __syncthreads();
  const int n = blockIdx.x * 256 + threadIdx.x;
  const float x = cloud[n * 3 + 0], y = cloud[n * 3 + 1], z = cloud[n * 3 + 2];
  float h[64];
#pragma unroll
  for (int k = 0; k < 64; ++k)
    h[k] = lrelu(sw1[k * 3] * x + sw1[k * 3 + 1] * y + sw1[k * 3 + 2] * z + b1[k]);
  for (int c = 0; c < 128; ++c) {
    float acc = b2[c];
#pragma unroll
    for (int k = 0; k < 64; ++k) acc += sw2[c * 64 + k] * h[k];
    cf16[n * 128 + c] = (_Float16)acc;
    fbuf[n * 416 + 288 + c] = (_Float16)lrelu(acc);
  }
}

// -------- KNN: per-point top-16 smallest d2 via register insertion sort
__global__ void __launch_bounds__(256)
knn_kernel(const float* __restrict__ cloud, int* __restrict__ nbr, float* __restrict__ wgt) {
  __shared__ float sx[1024], sy[1024], sz[1024];
  const int n = blockIdx.x * 256 + threadIdx.x;
  const float px = cloud[n * 3 + 0], py = cloud[n * 3 + 1], pz = cloud[n * 3 + 2];
  float bd[NBK]; int bi[NBK];
#pragma unroll
  for (int q = 0; q < NBK; ++q) { bd[q] = 3.4e38f; bi[q] = 0; }
  for (int t0 = 0; t0 < NPT; t0 += 1024) {
    __syncthreads();
    for (int i = threadIdx.x; i < 1024; i += 256) {
      sx[i] = cloud[(t0 + i) * 3 + 0];
      sy[i] = cloud[(t0 + i) * 3 + 1];
      sz[i] = cloud[(t0 + i) * 3 + 2];
    }
    if (t0 + 1024 < NPT)  // prefetch next tile (global_prefetch_b8)
      __builtin_prefetch(&cloud[(t0 + 1024) * 3 + threadIdx.x * 12], 0, 1);
    __syncthreads();
    for (int i = 0; i < 1024; ++i) {
      float dx = px - sx[i], dy = py - sy[i], dz = pz - sz[i];
      float d = dx * dx + dy * dy + dz * dz;
      if (d < bd[NBK - 1]) {
        float nd = d; int ni = t0 + i;
#pragma unroll
        for (int q = 0; q < NBK; ++q) {
          if (nd < bd[q]) {
            float td = bd[q]; int ti = bi[q];
            bd[q] = nd; bi[q] = ni; nd = td; ni = ti;
          }
        }
      }
    }
  }
  float dist[NBK], s = 0.f;
#pragma unroll
  for (int q = 0; q < NBK; ++q) dist[q] = sqrtf(fmaxf(bd[q], 1e-12f));
  const float dmin = dist[0];
#pragma unroll
  for (int q = 0; q < NBK; ++q) { dist[q] = expf(dmin - dist[q]); s += dist[q]; }
  const float inv = 1.f / s;
#pragma unroll
  for (int q = 0; q < NBK; ++q) { nbr[n * NBK + q] = bi[q]; wgt[n * NBK + q] = dist[q] * inv; }
}

// -------- Fused neighbor convs: one wave32 per point, 4 points per wave.
// All packed B fragments staged global->LDS via async copy, amortized over 32 points/block.
#define PPW 4
__global__ void __launch_bounds__(256)
fused_nb(const _Float16* __restrict__ imf16, const _Float16* __restrict__ cf16,
         const int* __restrict__ nbr, const float* __restrict__ wgt,
         const _Float16* __restrict__ pw,
         const float* __restrict__ c1b, const float* __restrict__ c2b,
         const float* __restrict__ p1b, const float* __restrict__ p2b,
         _Float16* __restrict__ fbuf) {
  __shared__ __align__(16) _Float16 wlds[75776];     // 148 KB packed B1|B2|B3|B4
  __shared__ __align__(16) _Float16 scr[8 * 4096];   //  64 KB per-wave D->A scratch
  const int wave = threadIdx.x >> 5;
  const int lane = threadIdx.x & 31;
  const int j  = lane & 15;
  const int hi = lane >> 4;
  _Float16* S = scr + wave * 4096;

  // ---- async-stage all packed weights into LDS (9472 x 16B chunks) ----
  for (int i = threadIdx.x; i < 9472; i += 256)
    async_cp16((void*)(pw + i * 8), (void*)(wlds + i * 8));
  wait_async0();
  __syncthreads();

  const _Float16* B1 = wlds;              // conv1:   1 kc x  4 nt
  const _Float16* B2 = wlds + 2048;       // conv2:   2 kc x  8 nt
  const _Float16* B3 = wlds + 10240;      // psconv1: 4 kc x 16 nt
  const _Float16* B4 = wlds + 43008;      // psconv2: 8 kc x  8 nt

  const int pbase = (blockIdx.x * 8 + wave) * PPW;
  for (int it = 0; it < PPW; ++it) {
    const int p = pbase + it;
    const int   row = nbr[p * NBK + j];
    const float wj  = wgt[p * NBK + j];

    const _Float16* ib = imf16 + row * 32;
    v16h a_img = load_a2(ib + hi * 8, ib + 16 + hi * 8);
    v16h a_cf[4];
    const _Float16* cb = cf16 + row * 128;
#pragma unroll
    for (int kc = 0; kc < 4; ++kc)
      a_cf[kc] = load_a2(cb + kc * 32 + hi * 8, cb + kc * 32 + 16 + hi * 8);

    // ======== sf: stage 1 (K=32 -> 64 ch) ========
    for (int nt = 0; nt < 4; ++nt) {
      v8f acc = {};
      acc = wmma16(a_img, load_bfrag(B1 + nt * 512, lane), acc);
      float b = c1b[nt * 16 + j];
#pragma unroll
      for (int v = 0; v < 8; ++v)
        S[(v + 8 * hi) * 64 + nt * 16 + j] = (_Float16)lrelu(acc[v] + b);
    }
    sched_fence();
    // ======== sf: stage 2 (K=64 -> 128 ch) + weighted max-pool ========
    for (int nt = 0; nt < 8; ++nt) {
      v8f acc = {};
#pragma unroll
      for (int kc = 0; kc < 2; ++kc) {
        v16h a = load_a2(S + j * 64 + kc * 32 + hi * 8,
                         S + j * 64 + kc * 32 + 16 + hi * 8);
        acc = wmma16(a, load_bfrag(B2 + (kc * 8 + nt) * 512, lane), acc);
      }
      float b = c2b[nt * 16 + j];
      float pool = -3.4e38f;
#pragma unroll
      for (int v = 0; v < 8; ++v) {
        float wm = __shfl(wj, v + 8 * hi, 32);
        pool = fmaxf(pool, (acc[v] + b) * wm);
      }
      pool = fmaxf(pool, __shfl_xor(pool, 16, 32));
      if (hi == 0) fbuf[p * 416 + 160 + nt * 16 + j] = (_Float16)lrelu(pool);
    }
    sched_fence();

    // ======== sfp: stage 1 (K=128 -> 256 ch) ========
    for (int nt = 0; nt < 16; ++nt) {
      v8f acc = {};
#pragma unroll
      for (int kc = 0; kc < 4; ++kc)
        acc = wmma16(a_cf[kc], load_bfrag(B3 + (kc * 16 + nt) * 512, lane), acc);
      float b = p1b[nt * 16 + j];
#pragma unroll
      for (int v = 0; v < 8; ++v)
        S[(v + 8 * hi) * 256 + nt * 16 + j] = (_Float16)lrelu(acc[v] + b);
    }
    sched_fence();
    // ======== sfp: stage 2 (K=256 -> 128 ch) + weighted max-pool ========
    for (int nt = 0; nt < 8; ++nt) {
      v8f acc = {};
#pragma unroll
      for (int kc = 0; kc < 8; ++kc) {
        v16h a = load_a2(S + j * 256 + kc * 32 + hi * 8,
                         S + j * 256 + kc * 32 + 16 + hi * 8);
        acc = wmma16(a, load_bfrag(B4 + (kc * 8 + nt) * 512, lane), acc);
      }
      float b = p2b[nt * 16 + j];
      float pool = -3.4e38f;
#pragma unroll
      for (int v = 0; v < 8; ++v) {
        float wm = __shfl(wj, v + 8 * hi, 32);
        pool = fmaxf(pool, (acc[v] + b) * wm);
      }
      pool = fmaxf(pool, __shfl_xor(pool, 16, 32));
      if (hi == 0) fbuf[p * 416 + nt * 16 + j] = (_Float16)lrelu(pool);
    }
    sched_fence();
  }
}

// -------- final: [N,416] @ [416,32] + bias ; output transposed to [32,N]
__global__ void __launch_bounds__(256)
final_gemm(const _Float16* __restrict__ fbuf, const _Float16* __restrict__ B5,
           const float* __restrict__ fb, float* __restrict__ out) {
  const int wave = threadIdx.x >> 5, lane = threadIdx.x & 31;
  const int j = lane & 15, hi = lane >> 4;
  const int p0 = (blockIdx.x * 8 + wave) * 16;
  const _Float16* arow = fbuf + (p0 + j) * 416;
#pragma unroll
  for (int nt = 0; nt < 2; ++nt) {
    v8f acc = {};
    for (int kc = 0; kc < 13; ++kc) {
      v16h a = load_a2(arow + kc * 32 + hi * 8, arow + kc * 32 + 16 + hi * 8);
      acc = wmma16(a, load_bfrag(B5 + (kc * 2 + nt) * 512, lane), acc);
    }
    float b = fb[nt * 16 + j];
#pragma unroll
    for (int v = 0; v < 8; ++v)
      out[(nt * 16 + j) * NPT + p0 + v + 8 * hi] = acc[v] + b;
  }
}

extern "C" void kernel_launch(void* const* d_in, const int* in_sizes, int n_in,
                              void* d_out, int out_size, void* d_ws, size_t ws_size,
                              hipStream_t stream) {
  (void)in_sizes; (void)n_in; (void)out_size; (void)ws_size;
  const float* img  = (const float*)d_in[0];
  const float* cld  = (const float*)d_in[1];
  const float* c1w  = (const float*)d_in[2];
  const float* c1b  = (const float*)d_in[3];
  const float* c2w  = (const float*)d_in[4];
  const float* c2b  = (const float*)d_in[5];
  const float* p1w  = (const float*)d_in[6];
  const float* p1b  = (const float*)d_in[7];
  const float* p2w  = (const float*)d_in[8];
  const float* p2b  = (const float*)d_in[9];
  const float* q1w  = (const float*)d_in[10];
  const float* q1b  = (const float*)d_in[11];
  const float* q2w  = (const float*)d_in[12];
  const float* q2b  = (const float*)d_in[13];
  const float* fw   = (const float*)d_in[14];
  const float* fb   = (const float*)d_in[15];
  float* out = (float*)d_out;

  char* ws = (char*)d_ws;
  _Float16* imf16 = (_Float16*)(ws + 0);           //  1.0 MB
  _Float16* cf16  = (_Float16*)(ws + 1048576);     //  4.0 MB
  _Float16* fbuf  = (_Float16*)(ws + 5242880);     // 13.0 MB  [N][416] f16
  int*      nbr   = (int*)     (ws + 18874368);    //  1.0 MB
  float*    wgt   = (float*)   (ws + 19922944);    //  1.0 MB
  _Float16* pw    = (_Float16*)(ws + 20971520);    //  0.17 MB packed B fragments

  pack_w<<<8,   256, 0, stream>>>(c1w, pw + 0,     32,  4,  1);
  pack_w<<<32,  256, 0, stream>>>(c2w, pw + 2048,  64,  8,  2);
  pack_w<<<128, 256, 0, stream>>>(p1w, pw + 10240, 128, 16, 4);
  pack_w<<<128, 256, 0, stream>>>(p2w, pw + 43008, 256, 8,  8);
  pack_w<<<52,  256, 0, stream>>>(fw,  pw + 75776, 416, 2, 13);

  prep_imf  <<<64,  256, 0, stream>>>(img, imf16, fbuf);
  point_mlp <<<64,  256, 0, stream>>>(cld, q1w, q1b, q2w, q2b, cf16, fbuf);
  knn_kernel<<<64,  256, 0, stream>>>(cld, nbr, wgt);
  fused_nb  <<<512, 256, 0, stream>>>(imf16, cf16, nbr, wgt, pw, c1b, c2b, p1b, p2b, fbuf);
  final_gemm<<<128, 256, 0, stream>>>(fbuf, pw + 75776, fb, out);
}
